// FormulaNet_70712341561833
// MI455X (gfx1250) — matrix operand order repaired and checked by reading, not compiled, not run on hardware
//
#include <hip/hip_runtime.h>

typedef __attribute__((ext_vector_type(16))) __bf16       v16bf;
typedef __attribute__((ext_vector_type(8)))  float        v8f;
typedef __attribute__((ext_vector_type(4)))  unsigned int u32x4;

#define L_DIM      48
#define P_DIM      2048
#define D_DIM      50
#define AL_DIM     5
#define OH_DIM     255
#define DP         64      // padded hidden dim
#define KP_OH      256     // padded onehot K
#define AROW       384     // per-node A row: [c0|leaf|c1|c2|c3|c4], 64 each
#define NODES_BLK  64
#define THREADS    128

// workspace layout (bytes)
#define WS_OFF_BIAS 0u              // 7*64 f32
#define WS_OFF_B    4096u           // 45056 dwords of swizzled bf16 B fragments (176 KB)
#define WS_OFF_BUF  (256u * 1024u)  // L*P*64 bf16 level buffer (12 MB)

__device__ __forceinline__ unsigned short f2bf(float f) {
    union { float f; unsigned int u; } v; v.f = f;
    unsigned int u = v.u;
    u += 0x7fffu + ((u >> 16) & 1u);          // round-to-nearest-even
    return (unsigned short)(u >> 16);
}
__device__ __forceinline__ float bf2f(unsigned short h) {
    union { unsigned int u; float f; } v; v.u = ((unsigned int)h) << 16;
    return v.f;
}

// ---------------------------------------------------------------------------
// GEMM inner loop: one 16(node) x 64(out) tile per wave, bf16 WMMA, K = nk*32.
// A row per lane: node m = lane%16, K halves split by lane/16 (contiguous k =
// 16*half + s, same order as B). B is pre-swizzled fragment-major: frag = ko*4+ntile, 8 dwords/lane.
// ---------------------------------------------------------------------------
__device__ __forceinline__ void gemm_head(const unsigned short* __restrict__ aRow,
                                          const unsigned short* __restrict__ Bsw,
                                          int nk, int lane, int half, v8f acc[4]) {
    for (int ko = 0; ko < nk; ++ko) {
        const unsigned short* ap = aRow + ko * 32 + half * 16;
        union { u32x4 q[2]; v16bf v; } A;
        A.q[0] = *(const u32x4*)(ap);        // K = ko*32 + half*16 + 0..7
        A.q[1] = *(const u32x4*)(ap + 8);    // K = ko*32 + half*16 + 8..15
        for (int nt = 0; nt < 4; ++nt) {
            const unsigned short* bp = Bsw + (ko * 4 + nt) * 512 + lane * 16;
            union { u32x4 q[2]; v16bf v; } B;
            B.q[0] = ((const u32x4*)bp)[0];
            B.q[1] = ((const u32x4*)bp)[1];
            acc[nt] = __builtin_amdgcn_wmma_f32_16x16x32_bf16(
                false, A.v, false, B.v, (short)0, acc[nt], false, false);
        }
    }
}

// ---------------------------------------------------------------------------
// Weight prep: pad each head's W^T to (Kp x 64) bf16 and store fragment-major
// in WMMA B layout (lane L: n = nt*16 + L%16, K = ko*32 + (L/16)*16 + 2i+{0,1}).
// Head order: 0 leaf, 1 unary, 2 exist, 3 univ, 4 bin, 5 fun, 6 pred.
// fun/pred K-block permutation maps A layout [c0|leaf|c1..c4] onto
// W columns [leaf|ce0..ce4].
// ---------------------------------------------------------------------------
__global__ void prep_B(const float* __restrict__ W_leaf,  const float* __restrict__ W_unary,
                       const float* __restrict__ W_exist, const float* __restrict__ W_univ,
                       const float* __restrict__ W_bin,   const float* __restrict__ W_fun,
                       const float* __restrict__ W_pred,  unsigned int* __restrict__ Bsw) {
    int gid = blockIdx.x * blockDim.x + threadIdx.x;
    if (gid >= 45056) return;
    const int cumdw[8] = {0, 8192, 10240, 12288, 14336, 20480, 32768, 45056};
    int h = 0;
    while (gid >= cumdw[h + 1]) ++h;
    int local = gid - cumdw[h];
    int frag = local >> 8, ld = local & 255;
    int lane = ld >> 3,   i  = ld & 7;
    int ko = frag >> 2,   nt = frag & 3;
    int n  = nt * 16 + (lane & 15);
    int k0 = ko * 32 + (lane >> 4) * 16 + 2 * i;
    const float* W[7] = {W_leaf, W_unary, W_exist, W_univ, W_bin, W_fun, W_pred};
    const int Korig[7] = {255, 50, 50, 50, 150, 300, 300};
    float e[2];
    for (int q = 0; q < 2; ++q) {
        int k = k0 + q;
        float v = 0.f;
        if (n < D_DIM) {
            if (h == 0) {
                if (k < OH_DIM) v = W[0][n * OH_DIM + k];
            } else {
                int b = k >> 6, j = k & 63;
                if (j < D_DIM) {
                    int cb = b;
                    if (h >= 5) cb = (b == 0) ? 1 : ((b == 1) ? 0 : b);
                    v = W[h][n * Korig[h] + cb * D_DIM + j];
                }
            }
        }
        e[q] = v;
    }
    Bsw[gid] = ((unsigned int)f2bf(e[1]) << 16) | (unsigned int)f2bf(e[0]);
}

__global__ void prep_bias(const float* __restrict__ b_leaf, const float* __restrict__ b_unary,
                          const float* __restrict__ b_exist, const float* __restrict__ b_univ,
                          const float* __restrict__ b_bin, const float* __restrict__ b_fun,
                          const float* __restrict__ b_pred, float* __restrict__ biasP) {
    int tid = blockIdx.x * blockDim.x + threadIdx.x;
    if (tid >= 7 * DP) return;
    int h = tid / DP, n = tid % DP;
    const float* src[7] = {b_leaf, b_unary, b_exist, b_univ, b_bin, b_fun, b_pred};
    biasP[tid] = (n < D_DIM) ? src[h][n] : 0.f;
}

// ---------------------------------------------------------------------------
// One level: stage onehot (bf16) + gathered children into LDS, run leaf GEMM,
// feed leaf back into the per-node A row, run the 6 remaining heads, then
// branchlessly select per node_type and write bf16 embeddings to the buffer.
// ---------------------------------------------------------------------------
__global__ __launch_bounds__(THREADS) void level_kernel(
    int l,
    const int*   __restrict__ node_type,
    const int*   __restrict__ children,
    const float* __restrict__ onehot,
    const unsigned short* __restrict__ Bsw,
    const float* __restrict__ biasP,
    unsigned short* __restrict__ buf) {

    __shared__ __align__(16) unsigned short sh_oh[NODES_BLK * KP_OH]; // 32 KB
    __shared__ __align__(16) unsigned short sh_A [NODES_BLK * AROW];  // 48 KB
    __shared__ int sh_ty[NODES_BLK];

    const int t    = threadIdx.x;
    const int blk0 = blockIdx.x * NODES_BLK;

    if (t < NODES_BLK) sh_ty[t] = node_type[l * P_DIM + blk0 + t];

    // stage onehot, f32 -> bf16, padded 255 -> 256
    for (int idx = t; idx < NODES_BLK * 128; idx += THREADS) {
        int nl = idx >> 7, pair = idx & 127, k = pair * 2;
        const float* src = onehot + (size_t)(l * P_DIM + blk0 + nl) * OH_DIM;
        float a0 = (k     < OH_DIM) ? src[k]     : 0.f;
        float a1 = (k + 1 < OH_DIM) ? src[k + 1] : 0.f;
        ((unsigned int*)sh_oh)[nl * 128 + pair] =
            ((unsigned int)f2bf(a1) << 16) | (unsigned int)f2bf(a0);
    }

    // gather child embeddings (bf16 dwords) into A-row c-blocks
    const unsigned int* bufd = (const unsigned int*)buf;
    for (int idx = t; idx < NODES_BLK * 160; idx += THREADS) {
        int nl = idx / 160, rem = idx - nl * 160;
        int j = rem >> 5, d = rem & 31;
        int c = children[(size_t)(l * P_DIM + blk0 + nl) * AL_DIM + j];
        unsigned int val = 0;
        if (c >= 0) val = bufd[(size_t)c * 32 + d];
        int dst = (j == 0) ? d : (64 + (j - 1) * 32 + d);   // c0|_|c1|c2|c3|c4
        ((unsigned int*)sh_A)[nl * 192 + dst] = val;
    }

    __syncthreads();

    const int lane     = t & 31;
    const int w        = t >> 5;
    const int half     = lane >> 4;
    const int m        = lane & 15;
    const int node_loc = w * 16 + m;

    int ty[8];
    for (int r = 0; r < 8; ++r) ty[r] = sh_ty[w * 16 + half * 8 + r];

    const int cumdw[8] = {0, 8192, 10240, 12288, 14336, 20480, 32768, 45056};
    const int nkArr[7] = {8, 2, 2, 2, 6, 12, 12};

    v8f sel[4];
    for (int nt = 0; nt < 4; ++nt)
        for (int r = 0; r < 8; ++r) sel[nt][r] = 0.f;

    // head 0: leaf = relu(onehot @ W_leaf^T + b)
    {
        v8f acc[4];
        for (int nt = 0; nt < 4; ++nt)
            for (int r = 0; r < 8; ++r) acc[nt][r] = 0.f;
        gemm_head(sh_oh + node_loc * KP_OH, Bsw + cumdw[0] * 2, 8, lane, half, acc);
        for (int nt = 0; nt < 4; ++nt) {
            int n = nt * 16 + m;
            float b = biasP[0 * DP + n];
            for (int r = 0; r < 8; ++r) {
                float v = acc[nt][r] + b;
                v = v > 0.f ? v : 0.f;
                int tl = half * 8 + r;
                sh_A[(w * 16 + tl) * AROW + 64 + n] = f2bf(v);  // leaf block
                if (ty[r] == 0) sel[nt][r] = v;
            }
        }
    }

    __syncthreads();   // leaf block visible before fun/pred/bin read it

    for (int h = 1; h < 7; ++h) {
        v8f acc[4];
        for (int nt = 0; nt < 4; ++nt)
            for (int r = 0; r < 8; ++r) acc[nt][r] = 0.f;
        gemm_head(sh_A + node_loc * AROW, Bsw + cumdw[h] * 2, nkArr[h], lane, half, acc);
        for (int nt = 0; nt < 4; ++nt) {
            int n = nt * 16 + m;
            float b = biasP[h * DP + n];
            for (int r = 0; r < 8; ++r) {
                float v = acc[nt][r] + b;
                v = v > 0.f ? v : 0.f;
                sel[nt][r] = (ty[r] == h) ? v : sel[nt][r];
            }
        }
    }

    // write selected embeddings to bf16 level buffer
    for (int nt = 0; nt < 4; ++nt) {
        int n = nt * 16 + m;
        for (int r = 0; r < 8; ++r) {
            int tl = half * 8 + r;
            size_t row = (size_t)l * P_DIM + blk0 + w * 16 + tl;
            buf[row * DP + n] = f2bf(sel[nt][r]);
        }
    }
}

// final: out[255] = root(bf16,50) @ W_final^T + b_final
__global__ void final_kernel(const unsigned short* __restrict__ buf,
                             const float* __restrict__ W_final,
                             const float* __restrict__ b_final,
                             float* __restrict__ out) {
    int n = blockIdx.x * blockDim.x + threadIdx.x;
    if (n >= OH_DIM) return;
    const unsigned short* root = buf + ((size_t)L_DIM * P_DIM - 1) * DP;
    float acc = b_final[n];
    for (int k = 0; k < D_DIM; ++k)
        acc += bf2f(root[k]) * W_final[n * D_DIM + k];
    out[n] = acc;
}

extern "C" void kernel_launch(void* const* d_in, const int* in_sizes, int n_in,
                              void* d_out, int out_size, void* d_ws, size_t ws_size,
                              hipStream_t stream) {
    const int*   node_type = (const int*)d_in[0];
    const int*   children  = (const int*)d_in[1];
    const float* onehot    = (const float*)d_in[2];
    const float* W_leaf  = (const float*)d_in[3];  const float* b_leaf  = (const float*)d_in[4];
    const float* W_unary = (const float*)d_in[5];  const float* b_unary = (const float*)d_in[6];
    const float* W_exist = (const float*)d_in[7];  const float* b_exist = (const float*)d_in[8];
    const float* W_univ  = (const float*)d_in[9];  const float* b_univ  = (const float*)d_in[10];
    const float* W_bin   = (const float*)d_in[11]; const float* b_bin   = (const float*)d_in[12];
    const float* W_fun   = (const float*)d_in[13]; const float* b_fun   = (const float*)d_in[14];
    const float* W_pred  = (const float*)d_in[15]; const float* b_pred  = (const float*)d_in[16];
    const float* W_final = (const float*)d_in[17]; const float* b_final = (const float*)d_in[18];

    char* ws = (char*)d_ws;
    float*          biasP = (float*)(ws + WS_OFF_BIAS);
    unsigned int*   BswD  = (unsigned int*)(ws + WS_OFF_B);
    unsigned short* Bsw   = (unsigned short*)(ws + WS_OFF_B);
    unsigned short* buf   = (unsigned short*)(ws + WS_OFF_BUF);

    prep_bias<<<2, 256, 0, stream>>>(b_leaf, b_unary, b_exist, b_univ, b_bin, b_fun, b_pred, biasP);
    prep_B<<<(45056 + 255) / 256, 256, 0, stream>>>(W_leaf, W_unary, W_exist, W_univ,
                                                    W_bin, W_fun, W_pred, BswD);
    for (int l = 0; l < L_DIM; ++l)
        level_kernel<<<P_DIM / NODES_BLK, THREADS, 0, stream>>>(
            l, node_type, children, onehot, Bsw, biasP, buf);
    final_kernel<<<1, 256, 0, stream>>>(buf, W_final, b_final, (float*)d_out);
}
